// GraMambaLayer_55980603736501
// MI455X (gfx1250) — compile-verified, matrix operand order checked by reading, not run on hardware
//
#include <hip/hip_runtime.h>
#include <hip/hip_bf16.h>
#include <cstddef>

// ---------------------------------------------------------------------------
// GraMamba layer for MI455X (gfx1250, wave32, WMMA).
// B=2 T=512 D_MODEL=768 D_STATE=16 D_CONV=4 D_INNER=1536
// GEMMs: bf16 WMMA, wave computes 2x2 = 32x32 tile (1 frag load per WMMA).
// ---------------------------------------------------------------------------

#define BB 2
#define TT 512
#define DM 768
#define DS 16
#define DC 4
#define DI 1536
#define ROWS (BB * TT)          // 1024
#define TPAD (TT + DC - 1)      // 515

typedef __attribute__((ext_vector_type(16))) __bf16 bf16x16;
typedef __attribute__((ext_vector_type(8)))  __bf16 bf16x8;
typedef __attribute__((ext_vector_type(8)))  float  f32x8;

// ---- fragment loader: two contiguous 16B loads per lane (per ISA layout) ----
__device__ __forceinline__ bf16x16 load_frag(const __hip_bfloat16* p) {
    bf16x16 f;
    ((bf16x8*)&f)[0] = *(const bf16x8*)(p);
    ((bf16x8*)&f)[1] = *(const bf16x8*)(p + 16);
    return f;
}

__device__ __forceinline__ f32x8 wmma_bf16(bf16x16 a, bf16x16 b, f32x8 c) {
    return __builtin_amdgcn_wmma_f32_16x16x32_bf16(
        false, a, false, b, (short)0, c, false, false);
}

__device__ __forceinline__ float silu_f(float v) {
    return v * (1.0f / (1.0f + __expf(-v)));
}

// ---------------------------------------------------------------------------
// 1. LayerNorm -> bf16 rows (one block per row)
// ---------------------------------------------------------------------------
__global__ __launch_bounds__(256) void ln_kernel(
    const float* __restrict__ x, const float* __restrict__ scale,
    const float* __restrict__ bias, __hip_bfloat16* __restrict__ xn) {
    const int row = blockIdx.x;
    const int tid = threadIdx.x;
    const float* xr = x + (size_t)row * DM;
    float s = 0.f, s2 = 0.f;
    for (int i = tid; i < DM; i += 256) { float v = xr[i]; s += v; s2 += v * v; }
    __shared__ float r1[256], r2[256];
    r1[tid] = s; r2[tid] = s2;
    __syncthreads();
    for (int off = 128; off > 0; off >>= 1) {
        if (tid < off) { r1[tid] += r1[tid + off]; r2[tid] += r2[tid + off]; }
        __syncthreads();
    }
    const float mu  = r1[0] * (1.0f / DM);
    const float var = r2[0] * (1.0f / DM) - mu * mu;
    const float inv = rsqrtf(var + 1e-5f);
    for (int i = tid; i < DM; i += 256)
        xn[(size_t)row * DM + i] = __float2bfloat16((xr[i] - mu) * inv * scale[i] + bias[i]);
}

// ---------------------------------------------------------------------------
// 2. generic f32 -> bf16 convert / weight reshapes
// ---------------------------------------------------------------------------
__global__ __launch_bounds__(256) void cvt_bf16_kernel(
    const float* __restrict__ src, __hip_bfloat16* __restrict__ dst, int n) {
    int i = blockIdx.x * 256 + threadIdx.x;
    if (i < n) dst[i] = __float2bfloat16(src[i]);
}

// conv_w [o][i][k] -> bf16 [k][o][i]
__global__ __launch_bounds__(256) void convw_perm_kernel(
    const float* __restrict__ src, __hip_bfloat16* __restrict__ dst) {
    int idx = blockIdx.x * 256 + threadIdx.x;
    if (idx >= DI * DI * DC) return;
    int k = idx & 3;
    int t = idx >> 2;
    int i = t % DI;
    int o = t / DI;
    dst[((size_t)k * DI + o) * DI + i] = __float2bfloat16(src[idx]);
}

// W_x (33,1536) -> WxT (1536,33) f32
__global__ __launch_bounds__(256) void wx_transpose_kernel(
    const float* __restrict__ src, float* __restrict__ dst) {
    int idx = blockIdx.x * 256 + threadIdx.x;
    if (idx >= 33 * DI) return;
    int j = idx / DI, i = idx % DI;
    dst[i * 33 + j] = src[idx];
}

// ---------------------------------------------------------------------------
// 3. Generic WMMA GEMM: C(M,N) = A(M,K) @ B(N,K)^T [+ resid], bf16 in, f32 out
//    block = 256 threads = 8 waves (2x4); wave = 32x32 tile (2x2 WMMA tiles)
//    -> block tile 64x128; 1 fragment load per WMMA (16 FLOP/byte issued).
// ---------------------------------------------------------------------------
template <int RESID>
__global__ __launch_bounds__(256) void gemm_wmma_kernel(
    const __hip_bfloat16* __restrict__ A, const __hip_bfloat16* __restrict__ Bm,
    float* __restrict__ C, const float* __restrict__ resid,
    int M, int N, int K) {
    const int lane = threadIdx.x & 31;
    const int wave = threadIdx.x >> 5;
    const int m0 = blockIdx.x * 64 + (wave >> 2) * 32;
    const int n0 = blockIdx.y * 128 + (wave & 3) * 32;
    const int rc = lane & 15, hi = lane >> 4;

    const __hip_bfloat16* arow0 = A  + (size_t)(m0 + rc)      * K + hi * 8;
    const __hip_bfloat16* arow1 = A  + (size_t)(m0 + 16 + rc) * K + hi * 8;
    const __hip_bfloat16* brow0 = Bm + (size_t)(n0 + rc)      * K + hi * 8;
    const __hip_bfloat16* brow1 = Bm + (size_t)(n0 + 16 + rc) * K + hi * 8;

    f32x8 acc00 = {0.f,0.f,0.f,0.f,0.f,0.f,0.f,0.f};
    f32x8 acc01 = acc00, acc10 = acc00, acc11 = acc00;

    for (int kk = 0; kk < K; kk += 32) {
        __builtin_prefetch(arow0 + kk + 256, 0, 0);   // global_prefetch_b8
        bf16x16 a0 = load_frag(arow0 + kk);
        bf16x16 a1 = load_frag(arow1 + kk);
        bf16x16 b0 = load_frag(brow0 + kk);
        bf16x16 b1 = load_frag(brow1 + kk);
        acc00 = wmma_bf16(a0, b0, acc00);
        acc01 = wmma_bf16(a0, b1, acc01);
        acc10 = wmma_bf16(a1, b0, acc10);
        acc11 = wmma_bf16(a1, b1, acc11);
    }

    f32x8 accs[2][2] = {{acc00, acc01}, {acc10, acc11}};
#pragma unroll
    for (int ti = 0; ti < 2; ++ti)
#pragma unroll
        for (int tj = 0; tj < 2; ++tj) {
            const int col = n0 + tj * 16 + rc;
#pragma unroll
            for (int r = 0; r < 8; ++r) {
                size_t idx = (size_t)(m0 + ti * 16 + r + 8 * hi) * N + col;
                float v = accs[ti][tj][r];
                if (RESID) v += resid[idx];
                C[idx] = v;
            }
        }
}

// ---------------------------------------------------------------------------
// 4. split xz -> causal-padded bf16 x_proj: xpad[b][3+t][:] = xz[b,t,0:1536]
// ---------------------------------------------------------------------------
__global__ __launch_bounds__(256) void split_pad_kernel(
    const float* __restrict__ xz, __hip_bfloat16* __restrict__ xpad) {
    int idx = blockIdx.x * 256 + threadIdx.x;
    if (idx >= BB * TPAD * DI) return;
    int d = idx % DI;
    int r = idx / DI;
    int b = r / TPAD, t = r % TPAD;
    float v = (t < DC - 1) ? 0.f : xz[((size_t)(b * TT + (t - (DC - 1)))) * (2 * DI) + d];
    xpad[idx] = __float2bfloat16(v);
}

// ---------------------------------------------------------------------------
// 5. Conv-as-GEMM with WMMA: xc[b,t,o] = sum_k sum_i xpad[b,t+k,i]*w[k][o][i]
//    Same 2x2 wave tiling; epilogue x_act = silu(xc + conv_b).
// ---------------------------------------------------------------------------
__global__ __launch_bounds__(256) void conv_wmma_kernel(
    const __hip_bfloat16* __restrict__ xpad, const __hip_bfloat16* __restrict__ wkoi,
    const float* __restrict__ convb, float* __restrict__ xact) {
    const int lane = threadIdx.x & 31;
    const int wave = threadIdx.x >> 5;
    const int m0 = blockIdx.x * 64 + (wave >> 2) * 32;
    const int n0 = blockIdx.y * 128 + (wave & 3) * 32;
    const int rc = lane & 15, hi = lane >> 4;

    // 64-row block tiles never straddle the batch boundary (T=512 % 64 == 0)
    const int rowA0 = m0 + rc, rowA1 = m0 + 16 + rc;
    const int b0 = rowA0 >> 9, t0 = rowA0 & (TT - 1);
    const int b1 = rowA1 >> 9, t1 = rowA1 & (TT - 1);
    const __hip_bfloat16* arow0 = xpad + ((size_t)(b0 * TPAD + t0)) * DI + hi * 8;
    const __hip_bfloat16* arow1 = xpad + ((size_t)(b1 * TPAD + t1)) * DI + hi * 8;
    const __hip_bfloat16* brow0 = wkoi + (size_t)(n0 + rc)      * DI + hi * 8;
    const __hip_bfloat16* brow1 = wkoi + (size_t)(n0 + 16 + rc) * DI + hi * 8;

    f32x8 acc00 = {0.f,0.f,0.f,0.f,0.f,0.f,0.f,0.f};
    f32x8 acc01 = acc00, acc10 = acc00, acc11 = acc00;

#pragma unroll
    for (int k = 0; k < DC; ++k) {
        const __hip_bfloat16* a0p = arow0 + (size_t)k * DI;
        const __hip_bfloat16* a1p = arow1 + (size_t)k * DI;
        const __hip_bfloat16* b0p = brow0 + (size_t)k * DI * DI;
        const __hip_bfloat16* b1p = brow1 + (size_t)k * DI * DI;
        for (int kk = 0; kk < DI; kk += 32) {
            __builtin_prefetch(b0p + kk + 256, 0, 0);
            bf16x16 a0 = load_frag(a0p + kk);
            bf16x16 a1 = load_frag(a1p + kk);
            bf16x16 b0 = load_frag(b0p + kk);
            bf16x16 b1 = load_frag(b1p + kk);
            acc00 = wmma_bf16(a0, b0, acc00);
            acc01 = wmma_bf16(a0, b1, acc01);
            acc10 = wmma_bf16(a1, b0, acc10);
            acc11 = wmma_bf16(a1, b1, acc11);
        }
    }

    f32x8 accs[2][2] = {{acc00, acc01}, {acc10, acc11}};
#pragma unroll
    for (int ti = 0; ti < 2; ++ti)
#pragma unroll
        for (int tj = 0; tj < 2; ++tj) {
            const int col = n0 + tj * 16 + rc;
            const float bias = convb[col];
#pragma unroll
            for (int r = 0; r < 8; ++r) {
                float v = accs[ti][tj][r] + bias;
                xact[(size_t)(m0 + ti * 16 + r + 8 * hi) * DI + col] = silu_f(v);
            }
        }
}

// ---------------------------------------------------------------------------
// 6. x_ssm = x_act @ W_x^T  (N = 33, tiny -> one wave per row, lane = out col)
// ---------------------------------------------------------------------------
__global__ __launch_bounds__(256) void xssm_kernel(
    const float* __restrict__ xact, const float* __restrict__ WxT,
    float* __restrict__ xssm) {
    const int row  = blockIdx.x * 8 + (threadIdx.x >> 5);
    const int lane = threadIdx.x & 31;
    if (row >= ROWS) return;
    const float* xr = xact + (size_t)row * DI;
    float a0 = 0.f, a1 = 0.f;
    for (int i = 0; i < DI; ++i) {
        float xv = xr[i];                       // broadcast load
        a0 += xv * WxT[i * 33 + lane];
        if (lane == 0) a1 += xv * WxT[i * 33 + 32];
    }
    xssm[(size_t)row * 33 + lane] = a0;
    if (lane == 0) xssm[(size_t)row * 33 + 32] = a1;
}

// ---------------------------------------------------------------------------
// 7. Sequential scan. One thread per channel d (state = 16 regs). Per step:
//    adjacency row packed to a 16-word bitmask in LDS; history gather streams
//    64B/channel/neighbor from L2 (hist = 100 MB < 192 MB L2).
// ---------------------------------------------------------------------------
__global__ __launch_bounds__(256) void scan_kernel(
    const float* __restrict__ xact, const float* __restrict__ xssm,
    const int*   __restrict__ adj,
    const float* __restrict__ Wdt, const float* __restrict__ bdt,
    const float* __restrict__ Wg,  const float* __restrict__ bg,
    float* __restrict__ hist, float* __restrict__ y) {
    const int b    = blockIdx.x / (DI / 256);
    const int dblk = blockIdx.x % (DI / 256);
    const int d    = dblk * 256 + threadIdx.x;

    __shared__ unsigned mask[16];
    __shared__ float    sWg[256 + 16];
    sWg[threadIdx.x] = Wg[threadIdx.x];                 // 16x16
    if (threadIdx.x < 16) sWg[256 + threadIdx.x] = bg[threadIdx.x];

    float h[DS];
#pragma unroll
    for (int n = 0; n < DS; ++n) h[n] = 0.f;
    const float wdt_d = Wdt[d];
    const float bdt_d = bdt[d];
    const float* histb = hist + (size_t)b * TT * DI * DS;

    for (int t = 0; t < TT; ++t) {
        __syncthreads();                                // protect mask reuse
        if (threadIdx.x < 16) {
            unsigned wbits = 0u;
            const int s0 = threadIdx.x * 32;
            const int smax = min(t, s0 + 32);           // strict lower triangle
            const int* arow = adj + ((size_t)b * TT + t) * TT;
            for (int s = s0; s < smax; ++s)
                if (arow[s] != 0) wbits |= 1u << (s - s0);
            mask[threadIdx.x] = wbits;
        }
        __syncthreads();

        // gather g = sum over neighbors of my own channel's history
        float g[DS];
#pragma unroll
        for (int n = 0; n < DS; ++n) g[n] = 0.f;
        int deg = 0;
#pragma unroll 1
        for (int w = 0; w < 16; ++w) {
            unsigned bits = mask[w];
            deg += __popc(bits);
            while (bits) {
                const int s = w * 32 + (__ffs(bits) - 1);
                bits &= bits - 1;
                const float4* hp = (const float4*)(histb + ((size_t)s * DI + d) * DS);
#pragma unroll
                for (int q = 0; q < 4; ++q) {
                    float4 v = hp[q];
                    g[q * 4 + 0] += v.x; g[q * 4 + 1] += v.y;
                    g[q * 4 + 2] += v.z; g[q * 4 + 3] += v.w;
                }
            }
        }

        // state decay + input injection
        const float* xr = xssm + ((size_t)b * TT + t) * 33;   // [draw | B(16) | C(16)]
        const float pre   = xr[0] * wdt_d + bdt_d;
        const float delta = (pre > 20.f) ? pre : __logf(1.f + __expf(pre));
        const float decay = __expf(-delta);
        const float xa    = xact[((size_t)b * TT + t) * DI + d];
#pragma unroll
        for (int n = 0; n < DS; ++n) h[n] = h[n] * decay + xr[1 + n] * xa;

        if (deg > 0) {
            const float inv = 1.0f / (float)deg;        // deg >= 1 == clip
#pragma unroll
            for (int n = 0; n < DS; ++n) g[n] *= inv;
#pragma unroll
            for (int n = 0; n < DS; ++n) {
                float acc = sWg[256 + n];
#pragma unroll
                for (int m = 0; m < DS; ++m) acc += g[m] * sWg[n * 16 + m];
                h[n] += 0.1f * silu_f(acc);
            }
        }

        // commit history + output y_t[d] = <h, C_t>
        float* hw = hist + (((size_t)b * TT + t) * DI + d) * DS;
        float yacc = 0.f;
#pragma unroll
        for (int n = 0; n < DS; ++n) { hw[n] = h[n]; yacc += h[n] * xr[17 + n]; }
        y[((size_t)b * TT + t) * DI + d] = yacc;
    }
}

// ---------------------------------------------------------------------------
// 8. gate: gated = bf16( y * silu(z) ),  z = xz[:, 1536:3072]
// ---------------------------------------------------------------------------
__global__ __launch_bounds__(256) void gate_kernel(
    const float* __restrict__ y, const float* __restrict__ xz,
    __hip_bfloat16* __restrict__ gated) {
    int idx = blockIdx.x * 256 + threadIdx.x;
    if (idx >= ROWS * DI) return;
    int r = idx / DI, d = idx % DI;
    float z = xz[(size_t)r * (2 * DI) + DI + d];
    gated[idx] = __float2bfloat16(y[idx] * silu_f(z));
}

// ---------------------------------------------------------------------------
extern "C" void kernel_launch(void* const* d_in, const int* in_sizes, int n_in,
                              void* d_out, int out_size, void* d_ws, size_t ws_size,
                              hipStream_t stream) {
    const float* x        = (const float*)d_in[0];
    const int*   adj      = (const int*)  d_in[1];
    const float* ln_scale = (const float*)d_in[2];
    const float* ln_bias  = (const float*)d_in[3];
    const float* W_in     = (const float*)d_in[4];
    const float* conv_w   = (const float*)d_in[5];
    const float* conv_b   = (const float*)d_in[6];
    const float* W_x      = (const float*)d_in[7];
    const float* W_dt     = (const float*)d_in[8];
    const float* b_dt     = (const float*)d_in[9];
    const float* W_g      = (const float*)d_in[10];
    const float* b_g      = (const float*)d_in[11];
    const float* W_out    = (const float*)d_in[12];
    float* out = (float*)d_out;

    char* ws = (char*)d_ws;
    size_t off = 0;
    auto alloc = [&](size_t bytes) -> void* {
        void* p = ws + off;
        off = (off + bytes + 255) & ~(size_t)255;
        return p;
    };
    __hip_bfloat16* xn_bf   = (__hip_bfloat16*)alloc((size_t)ROWS * DM * 2);
    __hip_bfloat16* Win_bf  = (__hip_bfloat16*)alloc((size_t)2 * DI * DM * 2);
    float*          xz      = (float*)         alloc((size_t)ROWS * 2 * DI * 4);
    __hip_bfloat16* xpad    = (__hip_bfloat16*)alloc((size_t)BB * TPAD * DI * 2);
    __hip_bfloat16* wkoi    = (__hip_bfloat16*)alloc((size_t)DC * DI * DI * 2);
    float*          xact    = (float*)         alloc((size_t)ROWS * DI * 4);
    float*          WxT     = (float*)         alloc((size_t)DI * 33 * 4);
    float*          xssm    = (float*)         alloc((size_t)ROWS * 33 * 4);
    float*          hist    = (float*)         alloc((size_t)ROWS * DI * DS * 4);  // 100 MB
    float*          yb      = (float*)         alloc((size_t)ROWS * DI * 4);
    __hip_bfloat16* gated   = (__hip_bfloat16*)alloc((size_t)ROWS * DI * 2);
    __hip_bfloat16* Wout_bf = (__hip_bfloat16*)alloc((size_t)DM * DI * 2);

    // 1. LN -> bf16
    ln_kernel<<<ROWS, 256, 0, stream>>>(x, ln_scale, ln_bias, xn_bf);
    // 2. weight converts
    {
        int n = 2 * DI * DM;
        cvt_bf16_kernel<<<(n + 255) / 256, 256, 0, stream>>>(W_in, Win_bf, n);
    }
    {
        int n = DC * DI * DI;
        convw_perm_kernel<<<(n + 255) / 256, 256, 0, stream>>>(conv_w, wkoi);
    }
    {
        int n = DM * DI;
        cvt_bf16_kernel<<<(n + 255) / 256, 256, 0, stream>>>(W_out, Wout_bf, n);
    }
    {
        int n = 33 * DI;
        wx_transpose_kernel<<<(n + 255) / 256, 256, 0, stream>>>(W_x, WxT);
    }
    // 3. xz = xn @ W_in^T   (M=1024, N=3072, K=768)
    gemm_wmma_kernel<0><<<dim3(ROWS / 64, (2 * DI) / 128), 256, 0, stream>>>(
        xn_bf, Win_bf, xz, nullptr, ROWS, 2 * DI, DM);
    // 4. causal pad
    {
        int n = BB * TPAD * DI;
        split_pad_kernel<<<(n + 255) / 256, 256, 0, stream>>>(xz, xpad);
    }
    // 5. conv GEMM + silu  (M=1024, N=1536, K=4*1536)
    conv_wmma_kernel<<<dim3(ROWS / 64, DI / 128), 256, 0, stream>>>(
        xpad, wkoi, conv_b, xact);
    // 6. x_ssm projection
    xssm_kernel<<<(ROWS + 7) / 8, 256, 0, stream>>>(xact, WxT, xssm);
    // 7. scan (12 blocks, one thread per channel)
    scan_kernel<<<BB * (DI / 256), 256, 0, stream>>>(
        xact, xssm, adj, W_dt, b_dt, W_g, b_g, hist, yb);
    // 8. gate
    {
        int n = ROWS * DI;
        gate_kernel<<<(n + 255) / 256, 256, 0, stream>>>(yb, xz, gated);
    }
    // 9. out = gated @ W_out^T + residual  (M=1024, N=768, K=1536)
    gemm_wmma_kernel<1><<<dim3(ROWS / 64, DM / 128), 256, 0, stream>>>(
        gated, Wout_bf, out, x, ROWS, DM, DI);
}
